// gatRecommer_58256936403226
// MI455X (gfx1250) — matrix operand order repaired and checked by reading, not compiled, not run on hardware
//
#include <hip/hip_runtime.h>
#include <math.h>

typedef __attribute__((ext_vector_type(16))) __bf16       v16bf;
typedef __attribute__((ext_vector_type(8)))  float        v8f;
typedef __attribute__((ext_vector_type(4)))  unsigned int v4u;

#define NN    50000
#define INCH  128
#define HID   256
#define NHEAD 4
#define CPH   64
#define OUTC  32

// ---------- helpers ----------
__device__ __forceinline__ unsigned short f2bf(float f) {
  unsigned u = __float_as_uint(f);
  unsigned r = u + 0x7fffu + ((u >> 16) & 1u);   // round-to-nearest-even
  return (unsigned short)(r >> 16);
}
// monotonic float->uint key so atomicMax(uint) == float max
__device__ __forceinline__ unsigned fkey(float f) {
  unsigned u = __float_as_uint(f);
  return u ^ ((unsigned)(((int)u) >> 31) | 0x80000000u);
}
__device__ __forceinline__ float unfkey(unsigned k) {
  unsigned u = (k & 0x80000000u) ? (k ^ 0x80000000u) : ~k;
  return __uint_as_float(u);
}

// ---------- conversion kernels ----------
__global__ void cvt_f32_bf16(const float* __restrict__ in,
                             unsigned short* __restrict__ out, int n) {
  for (int i = blockIdx.x * blockDim.x + threadIdx.x; i < n;
       i += gridDim.x * blockDim.x)
    out[i] = f2bf(in[i]);
}

// W [K,Nc] row-major f32  ->  Wt [Nc,K] row-major bf16 (transpose for B frags)
__global__ void wt_bf16(const float* __restrict__ W,
                        unsigned short* __restrict__ Wt, int K, int Nc) {
  int total = K * Nc;
  for (int i = blockIdx.x * blockDim.x + threadIdx.x; i < total;
       i += gridDim.x * blockDim.x) {
    int k = i / Nc, n = i % Nc;
    Wt[(size_t)n * K + k] = f2bf(W[i]);
  }
}

// ---------- WMMA GEMM: C[M,Nc] = A[M,K] (bf16) x Bt[Nc,K]^T (bf16), f32 accum
// one wave -> 16 x (16*NT) tile; 4 waves / block over consecutive M tiles.
// Software-pipelined with two *named* fragment sets (all constant indices ->
// everything stays in VGPRs): loads for K-step i+1 issue before the WMMAs of
// step i, so the NT WMMAs run back-to-back under partial loadcnt waits.
template <int NT>
__global__ void gemm_bf16_wmma(const unsigned short* __restrict__ A,
                               const unsigned short* __restrict__ Bt,
                               float* __restrict__ C, int M, int K, int Nc) {
  const int lane = threadIdx.x & 31;
  const int wave = threadIdx.x >> 5;
  const int mt   = blockIdx.x * 4 + wave;
  if (mt * 16 >= M) return;                      // wave-uniform guard
  const int m0  = mt * 16;
  const int n0  = blockIdx.y * (16 * NT);
  const int lm  = lane & 15;
  const int hi  = lane >> 4;                     // 0: lanes 0-15, 1: 16-31
  const int kbA = hi * 8;                        // A frag K base per half
  const int kbB = hi * 8;                        // B frag K base per half

  union Frag { v16bf v; v4u q[2]; };
  Frag a0, a1, b0[NT], b1[NT];
  v8f acc[NT];
#pragma unroll
  for (int t = 0; t < NT; ++t)
#pragma unroll
    for (int r = 0; r < 8; ++r) acc[t][r] = 0.0f;

  // hoisted row base pointers
  const unsigned short* Arow = A + (size_t)(m0 + lm) * K + kbA;
  const unsigned short* Brow[NT];
#pragma unroll
  for (int t = 0; t < NT; ++t)
    Brow[t] = Bt + (size_t)(n0 + t * 16 + lm) * K + kbB;

  // 16-bit A 16x32 layout: lanes0-15 K=0..7 & 16..23, lanes16-31 K=8..15 & 24..31
  // 16-bit B 32x16 layout: lanes0-15 K=0..7 & 16..23, lanes16-31 K=8..15 & 24..31 (N = lane&15)
#define LOAD_STEP(aF, bF, k0)                                   \
  do {                                                          \
    aF.q[0] = *(const v4u*)(Arow + (k0));                       \
    aF.q[1] = *(const v4u*)(Arow + (k0) + 16);                  \
    _Pragma("unroll")                                           \
    for (int t = 0; t < NT; ++t) {                              \
      bF[t].q[0] = *(const v4u*)(Brow[t] + (k0));               \
      bF[t].q[1] = *(const v4u*)(Brow[t] + (k0) + 16);          \
    }                                                           \
  } while (0)

#define WMMA_STEP(aF, bF)                                       \
  do {                                                          \
    _Pragma("unroll")                                           \
    for (int t = 0; t < NT; ++t)                                \
      acc[t] = __builtin_amdgcn_wmma_f32_16x16x32_bf16(         \
          false, aF.v, false, bF[t].v, (short)0, acc[t],        \
          false, false);                                        \
  } while (0)

  const int nk = K >> 5;                         // K/32 steps; even (K=128/256)
  LOAD_STEP(a0, b0, 0);
  for (int i = 0; i < nk; i += 2) {
    LOAD_STEP(a1, b1, (i + 1) << 5);             // prefetch step i+1
    WMMA_STEP(a0, b0);                           // compute step i
    if (i + 2 < nk) LOAD_STEP(a0, b0, (i + 2) << 5);  // prefetch step i+2
    WMMA_STEP(a1, b1);                           // compute step i+1
  }
#undef LOAD_STEP
#undef WMMA_STEP

#pragma unroll
  for (int t = 0; t < NT; ++t) {
    int n = n0 + t * 16 + lm;
#pragma unroll
    for (int r = 0; r < 8; ++r) {                // C/D: VGPR r -> M=r (+8 hi half)
      int m = m0 + r + hi * 8;
      C[(size_t)m * Nc + n] = acc[t][r];
    }
  }
}

// ---------- per-node attention logits: asrc/adst[n,h] = <h[n,h,:], a[h,:]> ----
__global__ void node_att(const float* __restrict__ h, const float* __restrict__ a_s,
                         const float* __restrict__ a_d, float* __restrict__ asrc,
                         float* __restrict__ adst, int nH, int C) {
  int t = blockIdx.x * blockDim.x + threadIdx.x;
  if (t >= NN * nH) return;
  int n = t / nH, hh = t % nH;
  const float* hp = h + (size_t)n * nH * C + (size_t)hh * C;
  const float* as = a_s + hh * C;
  const float* ad = a_d + hh * C;
  float s = 0.f, d = 0.f;
  for (int c = 0; c < C; ++c) { float v = hp[c]; s += v * as[c]; d += v * ad[c]; }
  asrc[t] = s; adst[t] = d;
}

// ---------- fills ----------
__global__ void fill_u32(unsigned* __restrict__ p, unsigned v, int n) {
  int i = blockIdx.x * blockDim.x + threadIdx.x;
  if (i < n) p[i] = v;
}
__global__ void bias_init(float* __restrict__ out, const float* __restrict__ b,
                          int hc, int total) {
  int i = blockIdx.x * blockDim.x + threadIdx.x;
  if (i < total) out[i] = b[i % hc];
}

// ---------- edge pass 1: segment max of leaky-relu logits ----------
__global__ void edge_logit_max(const int* __restrict__ src, const int* __restrict__ dst,
                               const float* __restrict__ asrc, const float* __restrict__ adst,
                               unsigned* __restrict__ mkey, int Ein, int nH) {
  int t = blockIdx.x * blockDim.x + threadIdx.x;
  int total = (Ein + NN) * nH;
  if (t >= total) return;
  int e = t / nH, hh = t % nH;
  int s = (e < Ein) ? src[e] : (e - Ein);        // implicit self-loops
  int d = (e < Ein) ? dst[e] : (e - Ein);
  float x = asrc[s * nH + hh] + adst[d * nH + hh];
  x = x > 0.f ? x : 0.2f * x;                    // leaky_relu(0.2)
  atomicMax(mkey + d * nH + hh, fkey(x));
}

// ---------- edge pass 2: exp(e - max) and segment sum ----------
__global__ void edge_exp_sum(const int* __restrict__ src, const int* __restrict__ dst,
                             const float* __restrict__ asrc, const float* __restrict__ adst,
                             const unsigned* __restrict__ mkey, float* __restrict__ z,
                             float* __restrict__ eexp, int Ein, int nH) {
  int t = blockIdx.x * blockDim.x + threadIdx.x;
  int total = (Ein + NN) * nH;
  if (t >= total) return;
  int e = t / nH, hh = t % nH;
  int s = (e < Ein) ? src[e] : (e - Ein);
  int d = (e < Ein) ? dst[e] : (e - Ein);
  float x = asrc[s * nH + hh] + adst[d * nH + hh];
  x = x > 0.f ? x : 0.2f * x;
  float ex = __expf(x - unfkey(mkey[d * nH + hh]));
  eexp[t] = ex;
  atomicAdd(z + d * nH + hh, ex);
}

// ---------- edge pass 3: out[dst] += alpha * h[src]  (one wave per edge) ------
__global__ void edge_aggr(const int* __restrict__ src, const int* __restrict__ dst,
                          const float* __restrict__ h, const float* __restrict__ eexp,
                          const float* __restrict__ z, float* __restrict__ out,
                          int Ein, int nH, int C) {
  int lane = threadIdx.x & 31;
  int e = (blockIdx.x * blockDim.x + threadIdx.x) >> 5;
  int total = Ein + NN;
  if (e >= total) return;
  int s = (e < Ein) ? src[e] : (e - Ein);
  int d = (e < Ein) ? dst[e] : (e - Ein);
  int hc = nH * C;
  const float* hs = h + (size_t)s * hc;
  float* od = out + (size_t)d * hc;
  for (int c = lane; c < hc; c += 32) {          // coalesced gather + L2 atomics
    int hh = c / C;
    float alpha = eexp[(size_t)e * nH + hh] / (z[d * nH + hh] + 1e-16f);
    atomicAdd(od + c, hs[c] * alpha);
  }
}

// ---------- fused LayerNorm + ELU (one wave / node, 256 ch, 8 per lane) ------
__global__ void ln_elu(float* __restrict__ h, const float* __restrict__ g,
                       const float* __restrict__ be) {
  int lane = threadIdx.x & 31;
  int n = (blockIdx.x * blockDim.x + threadIdx.x) >> 5;
  if (n >= NN) return;
  float* row = h + (size_t)n * HID;
  float v[8];
  float s = 0.f;
#pragma unroll
  for (int i = 0; i < 8; ++i) { v[i] = row[lane + i * 32]; s += v[i]; }
#pragma unroll
  for (int off = 16; off; off >>= 1) s += __shfl_xor(s, off, 32);
  float mu = s * (1.0f / HID);
  float q = 0.f;
#pragma unroll
  for (int i = 0; i < 8; ++i) { float dd = v[i] - mu; q += dd * dd; }
#pragma unroll
  for (int off = 16; off; off >>= 1) q += __shfl_xor(q, off, 32);
  float rs = rsqrtf(q * (1.0f / HID) + 1e-5f);
#pragma unroll
  for (int i = 0; i < 8; ++i) {
    int c = lane + i * 32;
    float y = (v[i] - mu) * rs * g[c] + be[c];
    row[c] = y > 0.f ? y : (__expf(y) - 1.f);    // ELU
  }
}

// ================== host ==================
static inline unsigned g1d(long n, int b) { return (unsigned)((n + b - 1) / b); }

extern "C" void kernel_launch(void* const* d_in, const int* in_sizes, int n_in,
                              void* d_out, int out_size, void* d_ws, size_t ws_size,
                              hipStream_t stream) {
  const float* x   = (const float*)d_in[0];
  const int*   ei  = (const int*)d_in[1];
  const int Ein    = in_sizes[1] / 2;            // 800000
  const int* src   = ei;
  const int* dst   = ei + Ein;
  const float *W1 = (const float*)d_in[2],  *a1s = (const float*)d_in[3],
              *a1d = (const float*)d_in[4], *b1  = (const float*)d_in[5],
              *gg1 = (const float*)d_in[6], *be1 = (const float*)d_in[7],
              *W2 = (const float*)d_in[8],  *a2s = (const float*)d_in[9],
              *a2d = (const float*)d_in[10],*b2  = (const float*)d_in[11],
              *gg2 = (const float*)d_in[12],*be2 = (const float*)d_in[13],
              *W3 = (const float*)d_in[14], *a3s = (const float*)d_in[15],
              *a3d = (const float*)d_in[16],*b3  = (const float*)d_in[17];
  float* out = (float*)d_out;

  // ---- carve workspace ----
  char* w = (char*)d_ws;
  auto carve = [&](size_t bytes) { void* p = w; w += (bytes + 255) & ~(size_t)255; return p; };
  float*          hA   = (float*)carve((size_t)NN * HID * 4);        // GEMM out
  float*          hB   = (float*)carve((size_t)NN * HID * 4);        // aggr out / next in
  unsigned short* hbf  = (unsigned short*)carve((size_t)NN * HID * 2);
  unsigned short* Wt   = (unsigned short*)carve((size_t)HID * HID * 2);
  float*          asrc = (float*)carve((size_t)NN * NHEAD * 4);
  float*          adst = (float*)carve((size_t)NN * NHEAD * 4);
  unsigned*       mkey = (unsigned*)carve((size_t)NN * NHEAD * 4);
  float*          zsum = (float*)carve((size_t)NN * NHEAD * 4);
  float*          eexp = (float*)carve((size_t)(Ein + NN) * NHEAD * 4);

  const int ET = Ein + NN;
  const unsigned KEY_NEG_INF = 0x007FFFFFu;      // fkey(-inf)
  const dim3 B256(256), B128(128);
  const unsigned gM = g1d(NN / 16, 4);           // ceil(3125/4) blocks over M

  auto gat_layer = [&](const float* hin_f32, int inCh, const float* W,
                       const float* as_w, const float* ad_w, const float* bias,
                       int nH, int C, float* gemm_out, float* aggr_out) {
    int outCh = nH * C;
    // bf16 convert activations + transpose/convert weights
    cvt_f32_bf16<<<g1d((long)NN * inCh, 256), B256, 0, stream>>>(hin_f32, hbf, NN * inCh);
    wt_bf16<<<g1d((long)inCh * outCh, 256), B256, 0, stream>>>(W, Wt, inCh, outCh);
    // GEMM
    if (outCh == HID) {
      dim3 grid(gM, HID / 64);
      gemm_bf16_wmma<4><<<grid, B128, 0, stream>>>(hbf, Wt, gemm_out, NN, inCh, outCh);
    } else {                                      // outCh == 32
      dim3 grid(gM, 1);
      gemm_bf16_wmma<2><<<grid, B128, 0, stream>>>(hbf, Wt, gemm_out, NN, inCh, outCh);
    }
    // attention logits per node
    node_att<<<g1d((long)NN * nH, 256), B256, 0, stream>>>(gemm_out, as_w, ad_w,
                                                           asrc, adst, nH, C);
    // init segment buffers + bias-initialized output
    fill_u32<<<g1d((long)NN * nH, 256), B256, 0, stream>>>(mkey, KEY_NEG_INF, NN * nH);
    fill_u32<<<g1d((long)NN * nH, 256), B256, 0, stream>>>((unsigned*)zsum, 0u, NN * nH);
    bias_init<<<g1d((long)NN * outCh, 256), B256, 0, stream>>>(aggr_out, bias, outCh,
                                                               NN * outCh);
    // edge passes
    edge_logit_max<<<g1d((long)ET * nH, 256), B256, 0, stream>>>(src, dst, asrc, adst,
                                                                 mkey, Ein, nH);
    edge_exp_sum<<<g1d((long)ET * nH, 256), B256, 0, stream>>>(src, dst, asrc, adst,
                                                               mkey, zsum, eexp, Ein, nH);
    edge_aggr<<<g1d((long)ET * 32, 256), B256, 0, stream>>>(src, dst, gemm_out, eexp,
                                                            zsum, aggr_out, Ein, nH, C);
  };

  // ---- layer 1: x[50000,128] -> hB[50000,256], LN+ELU ----
  gat_layer(x, INCH, W1, a1s, a1d, b1, NHEAD, CPH, hA, hB);
  ln_elu<<<g1d((long)NN * 32, 256), B256, 0, stream>>>(hB, gg1, be1);

  // ---- layer 2: hB -> hB (gemm into hA, aggr back into hB), LN+ELU ----
  gat_layer(hB, HID, W2, a2s, a2d, b2, NHEAD, CPH, hA, hB);
  ln_elu<<<g1d((long)NN * 32, 256), B256, 0, stream>>>(hB, gg2, be2);

  // ---- layer 3: hB -> d_out[50000,32] (heads=1, no LN) ----
  gat_layer(hB, HID, W3, a3s, a3d, b3, 1, OUTC, hA, out);
}
